// WHT_expansion_34548716929183
// MI455X (gfx1250) — matrix operand order repaired and checked by reading, not compiled, not run on hardware
//
#include <hip/hip_runtime.h>
#include <hip/hip_bf16.h>
#include <math.h>

typedef __attribute__((ext_vector_type(16))) _Float16 v16h;
typedef __attribute__((ext_vector_type(8)))  _Float16 v8h;
typedef __attribute__((ext_vector_type(2)))  _Float16 h2;
typedef __attribute__((ext_vector_type(2)))  __fp16   f16x2;
typedef __attribute__((ext_vector_type(8)))  float    v8f;

#define IN_F  256
#define PAD_F 512
#define OUT_F 512

// Sylvester Hadamard entry: H[i][j] = (-1)^popcount(i & j)
__device__ __forceinline__ float hadsign(int a, int b) {
    return (__popc(a & b) & 1) ? -1.0f : 1.0f;
}

// Hardware tanh: single TRANS32 op on gfx1250 (vs ~30-op libm expansion).
__device__ __forceinline__ float fast_tanh(float x) {
#if __has_builtin(__builtin_amdgcn_tanhf)
    return __builtin_amdgcn_tanhf(x);
#elif __has_builtin(__builtin_amdgcn_tanh_f32)
    return __builtin_amdgcn_tanh_f32(x);
#else
    float r;
    asm("v_tanh_f32 %0, %1\n\tv_nop" : "=v"(r) : "v"(x));
    return r;
#endif
}

// Pack two f32 into packed f16 pair with one VALU op (v_cvt_pk_rtz_f16_f32).
__device__ __forceinline__ h2 pkrtz(float a, float b) {
    f16x2 t = __builtin_amdgcn_cvt_pkrtz(a, b);
    return __builtin_bit_cast(h2, t);
}

__global__ __launch_bounds__(256)
void wht_expansion_kernel(const float* __restrict__ x,
                          const float* __restrict__ T,
                          float* __restrict__ out,
                          int nrows)
{
    __shared__ _Float16 lds[8][PAD_F];   // 1 KB scratch per wave

    const int tid  = threadIdx.x;
    const int warp = tid >> 5;
    const int lane = tid & 31;
    const int hi   = lane >> 4;          // half-wave id (0/1)
    const int M    = lane & 15;          // row (A) / col (B,C,D) index
    const int kbA  = 8 * hi;             // A-operand K base for this lane
    const int kbB  = 16 * hi;            // B-operand K base for this lane

    // ---- constant WMMA fragments (registers, hoisted out of row loop) ----
    // H32 as B operand (32x16), tile t covers output cols 16t..16t+15
    v16h hB0, hB1;
#pragma unroll
    for (int e = 0; e < 16; ++e) {
        int k = (e < 8) ? kbA + e : 16 + kbA + (e - 8);
        hB0[e] = (_Float16)hadsign(k, M);
        hB1[e] = (_Float16)hadsign(k, M + 16);
    }
    // H16 as A operand (16x32), K=16..31 zero padding
    v16h hA;
#pragma unroll
    for (int e = 0; e < 16; ++e) {
        hA[e] = (_Float16)(hi == 0 ? hadsign(M, e) : 0.0f);
    }

    // ---- per-lane |T| in accumulator layout: feature = (r+8*hi)*32 + M + 16*t ----
    float aT[16];
#pragma unroll
    for (int t = 0; t < 2; ++t)
#pragma unroll
        for (int r = 0; r < 8; ++r)
            aT[t * 8 + r] = fabsf(T[(r + 8 * hi) * 32 + M + 16 * t]);

    const float S = 0.044194173824159216f;   // 2^(-9/2), one per WHT
    _Float16* my = lds[warp];
    const v8f cz = {};

    // Data fragments with loop-invariant zero regions: declared once so the
    // structural zeros (U rows 8..15 = feature pad; B-frag K=16..31) are not
    // re-materialized every iteration.
    v16h aU = {};
    v16h b0 = {}, b1 = {};
    v16h c0 = {}, c1 = {};

    const long long gw = (long long)blockIdx.x * 8 + warp;
    const long long nw = (long long)gridDim.x * 8;

    for (long long row = gw; row < nrows; row += nw) {
        const float* rowp = x + row * (long long)IN_F;

        // Prefetch next row's input (1 KB) behind this row's compute.
        if (row + nw < (long long)nrows)
            __builtin_prefetch(x + (row + nw) * (long long)IN_F + lane * 8, 0, 1);

        // ===== WHT #1, stage 1:  V = U @ H32 =====
        // A-fragment = U (16x32) straight from global; only rows M<8 are real data.
        if (M < 8) {
            const float4* p0 = (const float4*)(rowp + M * 32 + kbA);
            float4 q0 = p0[0];
            float4 q1 = p0[1];
            const float4* p1 = (const float4*)(rowp + M * 32 + 16 + kbA);
            float4 q2 = p1[0];
            float4 q3 = p1[1];
            h2* pu = (h2*)&aU;
            pu[0] = pkrtz(q0.x, q0.y);  pu[1] = pkrtz(q0.z, q0.w);
            pu[2] = pkrtz(q1.x, q1.y);  pu[3] = pkrtz(q1.z, q1.w);
            pu[4] = pkrtz(q2.x, q2.y);  pu[5] = pkrtz(q2.z, q2.w);
            pu[6] = pkrtz(q3.x, q3.y);  pu[7] = pkrtz(q3.z, q3.w);
        }

        v8f V0 = __builtin_amdgcn_wmma_f32_16x16x32_f16(false, aU, false, hB0, (short)0, cz, false, false);
        v8f V1 = __builtin_amdgcn_wmma_f32_16x16x32_f16(false, aU, false, hB1, (short)0, cz, false, false);

        // ===== WHT #1, stage 2:  F1 = H16 @ V =====
        // Accumulator already holds V column-per-lane; lane n<16 needs its own 8
        // rows + lane (n+16)'s 8 rows -> 16 lane-swaps, no LDS.
        float o0[8], o1[8];
#pragma unroll
        for (int r = 0; r < 8; ++r) {
            o0[r] = __shfl_xor(V0[r], 16, 32);
            o1[r] = __shfl_xor(V1[r], 16, 32);
        }
        if (hi == 0) {
            h2* pb0 = (h2*)&b0;
            h2* pb1 = (h2*)&b1;
#pragma unroll
            for (int i = 0; i < 4; ++i) {
                pb0[i]     = pkrtz(V0[2 * i], V0[2 * i + 1]);
                pb0[4 + i] = pkrtz(o0[2 * i], o0[2 * i + 1]);
                pb1[i]     = pkrtz(V1[2 * i], V1[2 * i + 1]);
                pb1[4 + i] = pkrtz(o1[2 * i], o1[2 * i + 1]);
            }
        }
        v8f F0 = __builtin_amdgcn_wmma_f32_16x16x32_f16(false, hA, false, b0, (short)0, cz, false, false);
        v8f F1 = __builtin_amdgcn_wmma_f32_16x16x32_f16(false, hA, false, b1, (short)0, cz, false, false);

        // ===== nonlinearity in accumulator layout =====
        float f3v[16];
#pragma unroll
        for (int t = 0; t < 2; ++t)
#pragma unroll
            for (int r = 0; r < 8; ++r) {
                float f1 = (t == 0 ? F0[r] : F1[r]) * S;
                float g  = fabsf(f1) - aT[t * 8 + r];
                float v  = fast_tanh(f1) * fmaxf(g, 0.0f);
                if (t == 0 && r == 0 && lane == 0) v = f1;   // DC passthrough
                f3v[t * 8 + r] = v;
            }

        // ===== transpose F3 through LDS (same-wave producer/consumer) =====
#pragma unroll
        for (int t = 0; t < 2; ++t)
#pragma unroll
            for (int r = 0; r < 8; ++r)
                my[(r + 8 * hi) * 32 + M + 16 * t] = (_Float16)f3v[t * 8 + r];

        v16h aF;
        {
            v8h lo = *(const v8h*)(my + M * 32 + kbA);
            v8h hh = *(const v8h*)(my + M * 32 + 16 + kbA);
#pragma unroll
            for (int e = 0; e < 8; ++e) { aF[e] = lo[e]; aF[8 + e] = hh[e]; }
        }

        // ===== WHT #2 =====
        v8f W0 = __builtin_amdgcn_wmma_f32_16x16x32_f16(false, aF, false, hB0, (short)0, cz, false, false);
        v8f W1 = __builtin_amdgcn_wmma_f32_16x16x32_f16(false, aF, false, hB1, (short)0, cz, false, false);

        float p0a[8], p1a[8];
#pragma unroll
        for (int r = 0; r < 8; ++r) {
            p0a[r] = __shfl_xor(W0[r], 16, 32);
            p1a[r] = __shfl_xor(W1[r], 16, 32);
        }
        if (hi == 0) {
            h2* pc0 = (h2*)&c0;
            h2* pc1 = (h2*)&c1;
#pragma unroll
            for (int i = 0; i < 4; ++i) {
                pc0[i]     = pkrtz(W0[2 * i], W0[2 * i + 1]);
                pc0[4 + i] = pkrtz(p0a[2 * i], p0a[2 * i + 1]);
                pc1[i]     = pkrtz(W1[2 * i], W1[2 * i + 1]);
                pc1[4 + i] = pkrtz(p1a[2 * i], p1a[2 * i + 1]);
            }
        }
        v8f G0 = __builtin_amdgcn_wmma_f32_16x16x32_f16(false, hA, false, c0, (short)0, cz, false, false);
        v8f G1 = __builtin_amdgcn_wmma_f32_16x16x32_f16(false, hA, false, c1, (short)0, cz, false, false);

        // ===== store: each b32 store covers two contiguous 64B segments =====
        float* orow = out + row * (long long)OUT_F;
#pragma unroll
        for (int t = 0; t < 2; ++t)
#pragma unroll
            for (int r = 0; r < 8; ++r)
                orow[(r + 8 * hi) * 32 + M + 16 * t] = (t == 0 ? G0[r] : G1[r]) * S;
    }
}

extern "C" void kernel_launch(void* const* d_in, const int* in_sizes, int n_in,
                              void* d_out, int out_size, void* d_ws, size_t ws_size,
                              hipStream_t stream) {
    const float* x = (const float*)d_in[0];
    const float* T = (const float*)d_in[1];
    float* out     = (float*)d_out;

    int nrows = in_sizes[0] / IN_F;      // 4*256*256 = 262144 rows
    int need  = (nrows + 7) / 8;         // 8 waves per block, 1 row per wave-iter
    int blocks = need < 2048 ? (need > 0 ? need : 1) : 2048;

    hipLaunchKernelGGL(wht_expansion_kernel, dim3(blocks), dim3(256), 0, stream,
                       x, T, out, nrows);
}
